// MultiHeadAttention_15891378995275
// MI455X (gfx1250) — matrix operand order repaired and checked by reading, not compiled, not run on hardware
//
#include <hip/hip_runtime.h>
#include <stdint.h>

typedef __bf16 bf16;
typedef __attribute__((ext_vector_type(16))) __bf16 bf16x16;
typedef __attribute__((ext_vector_type(8)))  float  floatx8;
typedef __attribute__((address_space(3)))    bf16   lds_bf16;

union FragAB { bf16x16 v; unsigned u[8]; };

__device__ __forceinline__ bf16 f2bf(float f) { return (bf16)f; }

// CDNA5 async global->LDS copy (ASYNCcnt-tracked, fire-and-forget).
__device__ __forceinline__ void async_b128(void* lds_ptr, const void* gptr) {
  unsigned lds_off = (unsigned)(uintptr_t)(lds_bf16*)lds_ptr;  // addrspacecast -> 32-bit LDS offset
  asm volatile("global_load_async_to_lds_b128 %0, %1, off"
               :: "v"(lds_off), "v"(gptr) : "memory");
}
__device__ __forceinline__ void wait_async0() {
#if __has_builtin(__builtin_amdgcn_s_wait_asynccnt)
  __builtin_amdgcn_s_wait_asynccnt(0);
#else
  asm volatile("s_wait_asynccnt 0" ::: "memory");
#endif
}

// ---------------------------------------------------------------------------
// Problem constants (reference: n=4, t=2048, d=1024, h=16, dh=64)
// ---------------------------------------------------------------------------
constexpr int GD  = 1024;   // model dim (K and N of projection GEMMs)
constexpr int GM  = 8192;   // n*t rows
constexpr int T   = 2048;   // sequence length (per head after raw reshape)
constexpr int DH  = 64;     // head dim
constexpr int NH  = 16;     // heads
constexpr int NB  = 4;      // batch

// ---------------------------------------------------------------------------
// GEMM: C[M,N] = A[M,K] * W[K,N] + bias, bf16 WMMA, f32 accumulate.
// 128x128 WG tile (8 waves x 16x128 strip), BK=32, double-buffered LDS:
// one barrier per K-step, next tile's global loads overlap current WMMAs.
// ---------------------------------------------------------------------------
constexpr int BK  = 32;
constexpr int LDA = BK + 8;   // 80B row stride: conflict-free, 16B aligned

template <bool A_IS_F32, bool OUT_F32>
__global__ __launch_bounds__(256)
void gemm_kernel(const void* __restrict__ Ain, const float* __restrict__ W,
                 const float* __restrict__ bias, bf16* __restrict__ outb,
                 float* __restrict__ outf) {
  __shared__ bf16 As[2][128][LDA];
  __shared__ bf16 Wt[2][128][LDA];   // transposed weight tile: Wt[buf][n][k]

  const int tid  = threadIdx.x;
  const int wave = tid >> 5;
  const int lane = tid & 31;
  const int l15  = lane & 15;
  const int half = lane >> 4;
  const int n0   = blockIdx.x * 128;
  const int m0   = blockIdx.y * 128;

  floatx8 acc[8];
#pragma unroll
  for (int ct = 0; ct < 8; ++ct) acc[ct] = (floatx8)0.0f;

  // cooperative-load mapping
  const int arow = tid >> 1;          // 0..127 (2 threads per A row)
  const int acb  = (tid & 1) * 16;    // 16 elements each
  const int wk_  = tid & 31;          // W tile row (k) 0..31
  const int wcg  = tid >> 5;          // W col group (16 cols each)

  auto stage = [&](int buf, int k0) {
    if (A_IS_F32) {
      const float* Af = (const float*)Ain + (size_t)(m0 + arow) * GD + k0 + acb;
#pragma unroll
      for (int i = 0; i < 4; ++i) {
        float4 t = *(const float4*)(Af + 4 * i);
        As[buf][arow][acb + 4 * i + 0] = f2bf(t.x);
        As[buf][arow][acb + 4 * i + 1] = f2bf(t.y);
        As[buf][arow][acb + 4 * i + 2] = f2bf(t.z);
        As[buf][arow][acb + 4 * i + 3] = f2bf(t.w);
      }
    } else {
      const bf16* Ab = (const bf16*)Ain + (size_t)(m0 + arow) * GD + k0 + acb;
      *(uint4*)&As[buf][arow][acb]     = *(const uint4*)Ab;
      *(uint4*)&As[buf][arow][acb + 8] = *(const uint4*)(Ab + 8);
    }
    const float* Wf = W + (size_t)(k0 + wk_) * GD + n0 + wcg * 16;
#pragma unroll
    for (int i = 0; i < 4; ++i) {
      float4 t = *(const float4*)(Wf + 4 * i);
      const int c = wcg * 16 + 4 * i;
      Wt[buf][c + 0][wk_] = f2bf(t.x);
      Wt[buf][c + 1][wk_] = f2bf(t.y);
      Wt[buf][c + 2][wk_] = f2bf(t.z);
      Wt[buf][c + 3][wk_] = f2bf(t.w);
    }
  };

  stage(0, 0);
  __syncthreads();

  constexpr int NK = GD / BK;
#pragma unroll 2
  for (int ks = 0; ks < NK; ++ks) {
    const int p = ks & 1;
    if (ks + 1 < NK) stage(p ^ 1, (ks + 1) * BK);   // overlaps with WMMAs below

    // A fragment (16x32, one row per lane, ISA pair layout)
    FragAB af;
    const int am = wave * 16 + l15;
#pragma unroll
    for (int v = 0; v < 8; ++v) {
      const int kb = ((v < 4) ? 0 : 16) + half * 8 + (v & 3) * 2;
      af.u[v] = *(const unsigned*)&As[p][am][kb];
    }
#pragma unroll
    for (int ct = 0; ct < 8; ++ct) {
      FragAB bf_;
      const int bn = ct * 16 + l15;
#pragma unroll
      for (int v = 0; v < 8; ++v) {
        const int kb = ((v < 4) ? 0 : 16) + half * 8 + (v & 3) * 2;
        bf_.u[v] = *(const unsigned*)&Wt[p][bn][kb];
      }
      acc[ct] = __builtin_amdgcn_wmma_f32_16x16x32_bf16(
          false, af.v, false, bf_.v, (short)0, acc[ct], false, false);
    }
    __syncthreads();
  }

  // epilogue: bias + store
#pragma unroll
  for (int ct = 0; ct < 8; ++ct) {
    const int col = n0 + ct * 16 + l15;
    const float bv = bias[col];
#pragma unroll
    for (int r = 0; r < 8; ++r) {
      const int row = m0 + wave * 16 + r + 8 * half;
      const float val = acc[ct][r] + bv;
      if (OUT_F32) outf[(size_t)row * GD + col] = val;
      else         outb[(size_t)row * GD + col] = f2bf(val);
    }
  }
}

// ---------------------------------------------------------------------------
// Flash attention per (batch*head). 128 queries / WG (16 per wave).
// K/V streamed 64 keys at a time via ASYNC global->LDS (double-buffered),
// online softmax, P@V via WMMA.
// ---------------------------------------------------------------------------
constexpr int LKV = DH + 8;   // 144B row stride: conflict-free, 16B aligned

__global__ __launch_bounds__(256)
void attn_kernel(const bf16* __restrict__ Q, const bf16* __restrict__ Kk,
                 const bf16* __restrict__ Vv, const int* __restrict__ mask,
                 bf16* __restrict__ Out) {
  __shared__ bf16  Ks[2][64][LKV];
  __shared__ bf16  Vs[2][64][LKV];
  __shared__ bf16  Ps[8][16][LKV];   // per-wave P / output repack buffer
  __shared__ float smadd[2][64];

  const int tid  = threadIdx.x;
  const int wave = tid >> 5;
  const int lane = tid & 31;
  const int l15  = lane & 15;
  const int half = lane >> 4;
  const int bh   = blockIdx.y;          // b*NH + h
  const int b    = bh >> 4;
  const int q0   = blockIdx.x * 128 + wave * 16;
  const size_t hbase = (size_t)bh * T * DH;

  // Q fragments in registers for the whole loop (dh=64 -> 2 K-chunks)
  FragAB qa[2];
  {
    const bf16* qrow = Q + hbase + (size_t)(q0 + l15) * DH;
#pragma unroll
    for (int c = 0; c < 2; ++c)
#pragma unroll
      for (int v = 0; v < 8; ++v) {
        const int kb = c * 32 + ((v < 4) ? 0 : 16) + half * 8 + (v & 3) * 2;
        qa[c].u[v] = *(const unsigned*)(qrow + kb);
      }
  }

  floatx8 o[4];
#pragma unroll
  for (int ot = 0; ot < 4; ++ot) o[ot] = (floatx8)0.0f;
  float mrow[8], lrow[8];
#pragma unroll
  for (int r = 0; r < 8; ++r) { mrow[r] = -1e30f; lrow[r] = 0.0f; }

  const float scale = 0.125f;   // 1/sqrt(64)

  const int krow = tid >> 2;          // 64 rows, 4 threads per row
  const int kcb  = (tid & 3) * 16;

  auto stage = [&](int buf, int j) {
    const bf16* ksrc = Kk + hbase + (size_t)(j + krow) * DH + kcb;
    async_b128(&Ks[buf][krow][kcb],     ksrc);
    async_b128(&Ks[buf][krow][kcb + 8], ksrc + 8);
    const bf16* vsrc = Vv + hbase + (size_t)(j + krow) * DH + kcb;
    async_b128(&Vs[buf][krow][kcb],     vsrc);
    async_b128(&Vs[buf][krow][kcb + 8], vsrc + 8);
    if (tid < 64) smadd[buf][tid] = (mask[b * T + j + tid] != 0) ? 0.0f : -1e30f;
  };

  stage(0, 0);
  wait_async0();
  __syncthreads();

  for (int j = 0; j < T; j += 64) {
    const int p = (j >> 6) & 1;
    if (j + 64 < T) stage(p ^ 1, j + 64);   // async: overlaps with compute below

    // ---- scores: S = Q * K^T (4 column tiles x 2 K-chunks) ----
    floatx8 s[4];
#pragma unroll
    for (int ct = 0; ct < 4; ++ct) {
      floatx8 accs = (floatx8)0.0f;
#pragma unroll
      for (int c = 0; c < 2; ++c) {
        FragAB kb_;
#pragma unroll
        for (int v = 0; v < 8; ++v) {
          const int kd = c * 32 + ((v < 4) ? 0 : 16) + half * 8 + (v & 3) * 2;
          kb_.u[v] = *(const unsigned*)&Ks[p][ct * 16 + l15][kd];
        }
        accs = __builtin_amdgcn_wmma_f32_16x16x32_bf16(
            false, qa[c].v, false, kb_.v, (short)0, accs, false, false);
      }
      const float madd = smadd[p][ct * 16 + l15];
#pragma unroll
      for (int r = 0; r < 8; ++r) s[ct][r] = accs[r] * scale + madd;
    }

    // ---- online softmax (rows live in 16-lane halves: shuffle masks 1,2,4,8) ----
    float corr[8];
#pragma unroll
    for (int r = 0; r < 8; ++r) {
      float m_ = fmaxf(fmaxf(s[0][r], s[1][r]), fmaxf(s[2][r], s[3][r]));
      m_ = fmaxf(m_, __shfl_xor(m_, 1, 32));
      m_ = fmaxf(m_, __shfl_xor(m_, 2, 32));
      m_ = fmaxf(m_, __shfl_xor(m_, 4, 32));
      m_ = fmaxf(m_, __shfl_xor(m_, 8, 32));
      const float mnew = fmaxf(mrow[r], m_);
      corr[r] = __expf(mrow[r] - mnew);
      mrow[r] = mnew;
    }
    float rsum[8];
#pragma unroll
    for (int r = 0; r < 8; ++r) rsum[r] = 0.0f;
#pragma unroll
    for (int ct = 0; ct < 4; ++ct)
#pragma unroll
      for (int r = 0; r < 8; ++r) {
        const float pr = __expf(s[ct][r] - mrow[r]);
        rsum[r] += pr;
        Ps[wave][r + 8 * half][ct * 16 + l15] = f2bf(pr);   // repack to A layout
      }
#pragma unroll
    for (int r = 0; r < 8; ++r) {
      float t_ = rsum[r];
      t_ += __shfl_xor(t_, 1, 32);
      t_ += __shfl_xor(t_, 2, 32);
      t_ += __shfl_xor(t_, 4, 32);
      t_ += __shfl_xor(t_, 8, 32);
      lrow[r] = lrow[r] * corr[r] + t_;
    }
#pragma unroll
    for (int ot = 0; ot < 4; ++ot)
#pragma unroll
      for (int r = 0; r < 8; ++r) o[ot][r] *= corr[r];

    // ---- O += P * V ----
    FragAB pa[2];
#pragma unroll
    for (int c = 0; c < 2; ++c)
#pragma unroll
      for (int v = 0; v < 8; ++v) {
        const int kb = c * 32 + ((v < 4) ? 0 : 16) + half * 8 + (v & 3) * 2;
        pa[c].u[v] = *(const unsigned*)&Ps[wave][l15][kb];
      }
#pragma unroll
    for (int ot = 0; ot < 4; ++ot) {
#pragma unroll
      for (int c = 0; c < 2; ++c) {
        FragAB vb;
#pragma unroll
        for (int v = 0; v < 8; ++v) {
          const int kk = c * 32 + ((v < 4) ? 0 : 16) + half * 8 + (v & 3) * 2;
          unsigned short lo, hi;   // K-pairs span two V rows -> two u16 loads
          __builtin_memcpy(&lo, &Vs[p][kk][ot * 16 + l15], 2);
          __builtin_memcpy(&hi, &Vs[p][kk + 1][ot * 16 + l15], 2);
          vb.u[v] = (unsigned)lo | ((unsigned)hi << 16);
        }
        o[ot] = __builtin_amdgcn_wmma_f32_16x16x32_bf16(
            false, pa[c].v, false, vb.v, (short)0, o[ot], false, false);
      }
    }

    wait_async0();     // next buffer's async K/V landed
    __syncthreads();   // all waves done with buffer p
  }

  // ---- epilogue: normalize, repack via wave-private LDS, coalesced b128 stores ----
#pragma unroll
  for (int ot = 0; ot < 4; ++ot)
#pragma unroll
    for (int r = 0; r < 8; ++r)
      Ps[wave][r + 8 * half][ot * 16 + l15] = f2bf(o[ot][r] / lrow[r]);
  {
    bf16* drow = Out + hbase + (size_t)(q0 + l15) * DH + half * 32;
    const bf16* srow = &Ps[wave][l15][half * 32];
#pragma unroll
    for (int i = 0; i < 4; ++i)
      *(uint4*)(drow + i * 8) = *(const uint4*)(srow + i * 8);
  }
}

// ---------------------------------------------------------------------------
// Launch
// ---------------------------------------------------------------------------
extern "C" void kernel_launch(void* const* d_in, const int* in_sizes, int n_in,
                              void* d_out, int out_size, void* d_ws, size_t ws_size,
                              hipStream_t stream) {
  (void)in_sizes; (void)n_in; (void)out_size; (void)ws_size;
  const float* x    = (const float*)d_in[0];
  const int*   mask = (const int*)d_in[1];
  const float* wq   = (const float*)d_in[2];
  const float* bq   = (const float*)d_in[3];
  const float* wk   = (const float*)d_in[4];
  const float* bk   = (const float*)d_in[5];
  const float* wv   = (const float*)d_in[6];
  const float* bv   = (const float*)d_in[7];
  const float* wo   = (const float*)d_in[8];
  const float* bo   = (const float*)d_in[9];

  const size_t mat = (size_t)GM * GD;        // 8,388,608 elements
  bf16* Qb = (bf16*)d_ws;
  bf16* Kb = Qb + mat;
  bf16* Vb = Kb + mat;
  bf16* Ab = Vb + mat;                        // attention output (bf16)

  const dim3 gg(GD / 128, GM / 128);          // (8, 64)
  const dim3 ga(T / 128, NB * NH);            // (16, 64)

  gemm_kernel<true,  false><<<gg, 256, 0, stream>>>((const void*)x,  wq, bq, Qb, nullptr);
  gemm_kernel<true,  false><<<gg, 256, 0, stream>>>((const void*)x,  wk, bk, Kb, nullptr);
  gemm_kernel<true,  false><<<gg, 256, 0, stream>>>((const void*)x,  wv, bv, Vb, nullptr);
  attn_kernel<<<ga, 256, 0, stream>>>(Qb, Kb, Vb, mask, Ab);
  gemm_kernel<false, true ><<<gg, 256, 0, stream>>>((const void*)Ab, wo, bo, nullptr, (float*)d_out);
}
